// SPDetector_54460185313500
// MI455X (gfx1250) — matrix operand order repaired and checked by reading, not compiled, not run on hardware
//
#include <hip/hip_runtime.h>
#include <hip/hip_bf16.h>

typedef __attribute__((ext_vector_type(16))) _Float16 v16h;
typedef __attribute__((ext_vector_type(8)))  _Float16 v8h;
typedef __attribute__((ext_vector_type(4)))  _Float16 v4h;
typedef __attribute__((ext_vector_type(8)))  float    v8f;

#define BATCH 16
#define CIN   128
#define CMID  256
#define HF    96
#define WF    96
#define NPIX  (HF * WF)       // 9216
#define K1    (CIN * 9)       // 1152
#define NCLS  65
#define MPAD  80              // 65 padded to 5 WMMA tiles
#define HI    768
#define WI    768
#define KTOP  4096
#define CANDCAP 8192
#define PW    98              // padded input width (96 + halo)
#define PADP  (PW * PW)       // 9604 elems per padded plane

union Frag128 { float4 f4[2]; v16h h; };
union H8      { float4 f4;    v8h  h; };
union H4      { float2 f2;    v4h  h; };

__device__ inline v16h load_frag2(const _Float16* p0, const _Float16* p1) {
    Frag128 u;
    u.f4[0] = *(const float4*)p0;   // ds_load_b128
    u.f4[1] = *(const float4*)p1;   // ds_load_b128
    return u.h;
}

// ---------------------------------------------------------------------------
// Prep kernels (one-time): f16 conversions; x gets a zero halo so the conv1
// im2col gather needs no bounds checks (no exec divergence in the hot loop).
// ---------------------------------------------------------------------------
__global__ __launch_bounds__(256) void sp_cvt_f32_to_f16(
    const float* __restrict__ src, _Float16* __restrict__ dst, int n4)
{
    int i = blockIdx.x * 256 + threadIdx.x;
    if (i >= n4) return;
    float4 v = *(const float4*)(src + (size_t)i * 4);
    H4 o;
    o.h[0] = (_Float16)v.x; o.h[1] = (_Float16)v.y;
    o.h[2] = (_Float16)v.z; o.h[3] = (_Float16)v.w;
    *(float2*)(dst + (size_t)i * 4) = o.f2;
}

// w2 [65][256] -> f16 [80][256] zero-padded rows 65..79
__global__ __launch_bounds__(256) void sp_cvt_w2(
    const float* __restrict__ w2, _Float16* __restrict__ w2h)
{
    int i = blockIdx.x * 256 + threadIdx.x;          // one float4 chunk
    if (i >= MPAD * CMID / 4) return;
    int m = (i * 4) >> 8;                            // /256
    H4 o;
    if (m < NCLS) {
        float4 v = *(const float4*)(w2 + (size_t)i * 4);
        o.h[0] = (_Float16)v.x; o.h[1] = (_Float16)v.y;
        o.h[2] = (_Float16)v.z; o.h[3] = (_Float16)v.w;
    } else {
        o.h[0] = o.h[1] = o.h[2] = o.h[3] = (_Float16)0.f;
    }
    *(float2*)(w2h + (size_t)i * 4) = o.f2;
}

// x [B][128][96][96] f32 -> xh [B][128][98][98] f16 with zero halo
__global__ __launch_bounds__(256) void sp_cvt_x_pad(
    const float* __restrict__ x, _Float16* __restrict__ xh)
{
    unsigned i = blockIdx.x * 256 + threadIdx.x;
    const unsigned total = (unsigned)BATCH * CIN * PADP;   // 19,668,992
    if (i >= total) return;
    unsigned col = i % PW;
    unsigned t   = i / PW;
    unsigned row = t % PW;
    unsigned plane = t / PW;                               // img*CIN + ci
    int iy = (int)row - 1, ix = (int)col - 1;
    _Float16 v = (_Float16)0.f;
    if ((unsigned)iy < (unsigned)HF && (unsigned)ix < (unsigned)WF)
        v = (_Float16)x[(size_t)plane * NPIX + iy * WF + ix];
    xh[i] = v;
}

// ---------------------------------------------------------------------------
// Kernel 1: 3x3 conv (128 -> 256) + ReLU, implicit GEMM on WMMA f16.
// Block 256 thr = 8 waves as 4(Mquad of 32 chans) x 2(N): 128 chan x 32 pix
// tile, 2 WMMA per wave per K-step. Double-buffered LDS: one barrier per
// K-step; next tile's global loads overlap the current tile's WMMAs.
// B-gather uses the padded xh plane + incremental (ci,rem) update: no
// divides and no bounds branches in the loop.
// ---------------------------------------------------------------------------
__global__ __launch_bounds__(256) void sp_conv1_wmma(
    const _Float16* __restrict__ xh, const _Float16* __restrict__ w1h,
    const float* __restrict__ b1, _Float16* __restrict__ h)
{
    __shared__ __align__(16) _Float16 As[2][128 * 40];  // weights [m][k], pad 40
    __shared__ __align__(16) _Float16 Bs[2][32 * 40];   // im2col  [n][k], pad 40

    const int pixBase = blockIdx.x * 32;
    const int mBase   = blockIdx.y * 128;
    const int img     = blockIdx.z;

    const int tid   = threadIdx.x;
    const int lane  = tid & 31;
    const int wave  = tid >> 5;
    const int mQuad = wave & 3;     // 0..3 -> 32 channels each
    const int nTile = wave >> 2;    // 0..1
    const int hk    = lane >> 4;
    const int l16   = lane & 15;

    // A-staging invariants: 2 b128 chunks per thread per K-step
    const int aRow  = tid >> 2, aCh = tid & 3;
    const int aOff0 = aRow * 40 + aCh * 8;
    const int aOff1 = (aRow + 64) * 40 + aCh * 8;
    const _Float16* aSrc0 = w1h + (size_t)(mBase + aRow) * K1 + aCh * 8;
    const _Float16* aSrc1 = w1h + (size_t)(mBase + aRow + 64) * K1 + aCh * 8;

    // B-staging invariants: 4 scalar f16 per thread per K-step
    int ldsOff[4], xbase[4], ciS[4], remS[4];
    #pragma unroll
    for (int j = 0; j < 4; ++j) {
        int i  = tid + j * 256;
        int nn = i >> 5, kk = i & 31;
        ldsOff[j] = nn * 40 + kk;
        int p = pixBase + nn;
        int y = p / WF, xx = p - y * WF;
        xbase[j] = img * CIN * PADP + y * PW + xx;   // padded-plane base
        ciS[j]  = kk / 9;
        remS[j] = kk - ciS[j] * 9;
    }

#define C1_STAGE_A(buf, k0)                                                  \
    do {                                                                     \
        *(float4*)&As[buf][aOff0] = *(const float4*)(aSrc0 + (k0));          \
        *(float4*)&As[buf][aOff1] = *(const float4*)(aSrc1 + (k0));          \
    } while (0)

#define C1_STAGE_B(buf)                                                      \
    do {                                                                     \
        _Pragma("unroll")                                                    \
        for (int j = 0; j < 4; ++j) {                                        \
            int kh = (remS[j] * 11) >> 5;          /* rem/3 for rem in 0..8 */\
            int kw = remS[j] - kh * 3;                                       \
            Bs[buf][ldsOff[j]] =                                             \
                xh[xbase[j] + ciS[j] * PADP + kh * PW + kw];                 \
            int r2 = remS[j] + 5;                  /* k0 += 32 == +5 mod 9 */\
            int ov = r2 >= 9;                                                \
            remS[j] = ov ? r2 - 9 : r2;                                      \
            ciS[j] += ov ? 4 : 3;                                            \
        }                                                                    \
    } while (0)

    v8f c0 = {0.f,0.f,0.f,0.f,0.f,0.f,0.f,0.f};
    v8f c1 = {0.f,0.f,0.f,0.f,0.f,0.f,0.f,0.f};

    C1_STAGE_A(0, 0);
    C1_STAGE_B(0);
    __syncthreads();

    for (int k0 = 0; k0 < K1; k0 += 32) {
        const int cur = (k0 >> 5) & 1;
        const int nxt = cur ^ 1;
        if (k0 + 32 < K1) {                         // prefetch next tile
            C1_STAGE_A(nxt, k0 + 32);
            C1_STAGE_B(nxt);
        }
        const _Float16* bp = &Bs[cur][(nTile * 16 + l16) * 40 + hk * 8];
        v16h bfr = load_frag2(bp, bp + 16);
        const _Float16* a0 = &As[cur][(mQuad * 32 + l16) * 40 + hk * 8];
        const _Float16* a1 = a0 + 16 * 40;
        v16h af0 = load_frag2(a0, a0 + 16);
        v16h af1 = load_frag2(a1, a1 + 16);
        c0 = __builtin_amdgcn_wmma_f32_16x16x32_f16(
                 false, af0, false, bfr, (short)0, c0, false, false);
        c1 = __builtin_amdgcn_wmma_f32_16x16x32_f16(
                 false, af1, false, bfr, (short)0, c1, false, false);
        __syncthreads();                            // single barrier / K-step
    }

    // Epilogue: bias + ReLU -> h[img][pix][chan], one b128 store per subtile
    const int n = pixBase + nTile * 16 + l16;
    #pragma unroll
    for (int sub = 0; sub < 2; ++sub) {
        int mrow = mBase + mQuad * 32 + sub * 16 + 8 * hk;  // 8 contiguous chans
        v8f& c = sub ? c1 : c0;
        H8 o;
        #pragma unroll
        for (int r = 0; r < 8; ++r) {
            float v = c[r] + b1[mrow + r];
            o.h[r] = (_Float16)(v > 0.f ? v : 0.f);
        }
        *(float4*)&h[((size_t)img * NPIX + n) * CMID + mrow] = o.f4;
    }
#undef C1_STAGE_A
#undef C1_STAGE_B
}

// ---------------------------------------------------------------------------
// Kernel 2: 1x1 conv (256 -> 65 padded 80) + softmax + pixel-shuffle.
// Block 160 thr = 5 waves x 16 pixels; all staging b128; double-buffered.
// ---------------------------------------------------------------------------
__global__ __launch_bounds__(160) void sp_conv2_softmax(
    const _Float16* __restrict__ h, const _Float16* __restrict__ w2h,
    const float* __restrict__ b2, float* __restrict__ heat)
{
    __shared__ __align__(16) _Float16 As[2][MPAD * 40];
    __shared__ __align__(16) _Float16 Bs[2][16 * 40];
    __shared__ float semi[MPAD * 17];

    const int pixBase = blockIdx.x * 16;
    const int img     = blockIdx.y;
    const int tid  = threadIdx.x;
    const int lane = tid & 31;
    const int wave = tid >> 5;      // mTile 0..4
    const int hk   = lane >> 4;
    const int l16  = lane & 15;

    // A staging: 320 chunks, 2 per thread
    const int aRow0 = tid >> 2,        aCh0 = tid & 3;
    const int aRow1 = (tid + 160) >> 2, aCh1 = (tid + 160) & 3;
    const _Float16* aSrc0 = w2h + (size_t)aRow0 * CMID + aCh0 * 8;
    const _Float16* aSrc1 = w2h + (size_t)aRow1 * CMID + aCh1 * 8;
    const int aOff0 = aRow0 * 40 + aCh0 * 8;
    const int aOff1 = aRow1 * 40 + aCh1 * 8;
    // B staging: 64 chunks, waves 0-1 only (wave-uniform predicate)
    const int bRow = tid >> 2, bCh = tid & 3;
    const _Float16* bSrc =
        h + ((size_t)img * NPIX + pixBase + bRow) * CMID + bCh * 8;
    const int bOff = bRow * 40 + bCh * 8;

#define C2_STAGE(buf, k0)                                                    \
    do {                                                                     \
        *(float4*)&As[buf][aOff0] = *(const float4*)(aSrc0 + (k0));          \
        *(float4*)&As[buf][aOff1] = *(const float4*)(aSrc1 + (k0));          \
        if (tid < 64)                                                        \
            *(float4*)&Bs[buf][bOff] = *(const float4*)(bSrc + (k0));        \
    } while (0)

    v8f c = {0.f,0.f,0.f,0.f,0.f,0.f,0.f,0.f};

    C2_STAGE(0, 0);
    __syncthreads();

    for (int k0 = 0; k0 < CMID; k0 += 32) {
        const int cur = (k0 >> 5) & 1;
        const int nxt = cur ^ 1;
        if (k0 + 32 < CMID) C2_STAGE(nxt, k0 + 32);
        const _Float16* ap = &As[cur][(wave * 16 + l16) * 40 + hk * 8];
        v16h a = load_frag2(ap, ap + 16);
        const _Float16* bp = &Bs[cur][l16 * 40 + hk * 8];
        v16h bfr = load_frag2(bp, bp + 16);
        c = __builtin_amdgcn_wmma_f32_16x16x32_f16(
                false, a, false, bfr, (short)0, c, false, false);
        __syncthreads();
    }
#undef C2_STAGE

    #pragma unroll
    for (int r = 0; r < 8; ++r) {
        int m = wave * 16 + r + 8 * hk;
        float bias = (m < NCLS) ? b2[m] : 0.f;
        semi[m * 17 + l16] = c[r] + bias;
    }
    __syncthreads();

    if (tid < 16) {
        int p = pixBase + tid;
        int y = p / WF, xx = p - y * WF;
        float mx = -1e30f;
        for (int m = 0; m < NCLS; ++m) mx = fmaxf(mx, semi[m * 17 + tid]);
        float den = 0.f;
        for (int m = 0; m < NCLS; ++m) den += __expf(semi[m * 17 + tid] - mx);
        float inv = 1.f / den;
        for (int cls = 0; cls < 64; ++cls) {          // drop dust bin
            float pr = __expf(semi[cls * 17 + tid] - mx) * inv;
            int rr = cls >> 3, cc = cls & 7;
            heat[((size_t)img * HI + (y * 8 + rr)) * WI + (xx * 8 + cc)] = pr;
        }
    }
}

// ---------------------------------------------------------------------------
// Candidate gather + per-image top-K sort + greedy NMS.
// ---------------------------------------------------------------------------
__global__ void sp_zero_cnt(int* __restrict__ cnt)
{
    if (threadIdx.x < BATCH) cnt[threadIdx.x] = 0;
}

__global__ __launch_bounds__(256) void sp_collect(
    const float* __restrict__ heat, int* __restrict__ cnt,
    float2* __restrict__ cand)
{
    int img = blockIdx.y;
    int p = blockIdx.x * 256 + threadIdx.x;
    if (p >= HI * WI) return;
    float v = heat[(size_t)img * HI * WI + p];
    if (v >= 0.5f) {
        int slot = atomicAdd(&cnt[img], 1);
        if (slot < CANDCAP) cand[img * CANDCAP + slot] = make_float2(v, (float)p);
    }
}

__global__ __launch_bounds__(1024) void sp_nms(
    const float* __restrict__ heat, const int* __restrict__ cnt,
    const float2* __restrict__ cand, float* __restrict__ out)
{
    extern __shared__ unsigned char sm3[];
    float*         sc   = (float*)sm3;                        // 16384 B
    int*           id   = (int*)(sm3 + 16384);                // 16384 B
    unsigned*      mask = (unsigned*)(sm3 + 32768);           // 776*25*4 = 77600 B
    unsigned char* keep = sm3 + 32768 + 77600;                // 4096 B

    const int img = blockIdx.x;
    const int tid = threadIdx.x;
    const int n   = min(cnt[img], KTOP);

    for (int i = tid; i < KTOP; i += 1024) {
        if (i < n) { float2 ce = cand[img * CANDCAP + i]; sc[i] = ce.x; id[i] = (int)ce.y; }
        else       { sc[i] = -1.f; id[i] = 0; }
    }
    for (int w = tid; w < 776 * 25; w += 1024) {
        int prow = w / 25, wc = w - prow * 25;
        unsigned bits = 0u;
        int hr = prow - 4;
        if (hr >= 0 && hr < HI) {
            for (int bit = 0; bit < 32; ++bit) {
                int hc = wc * 32 + bit - 4;
                if (hc >= 0 && hc < WI &&
                    heat[((size_t)img * HI + hr) * WI + hc] >= 0.5f)
                    bits |= (1u << bit);
            }
        }
        mask[w] = bits;
    }
    __syncthreads();

    for (int ksz = 2; ksz <= KTOP; ksz <<= 1) {
        for (int j = ksz >> 1; j > 0; j >>= 1) {
            for (int i = tid; i < KTOP; i += 1024) {
                int ixj = i ^ j;
                if (ixj > i) {
                    bool up = (i & ksz) == 0;
                    float a = sc[i], b = sc[ixj];
                    bool sw = up ? (a < b) : (a > b);
                    if (sw) {
                        sc[i] = b; sc[ixj] = a;
                        int t = id[i]; id[i] = id[ixj]; id[ixj] = t;
                    }
                }
            }
            __syncthreads();
        }
    }

    if (tid == 0) {
        for (int j = 0; j < KTOP; ++j) {
            keep[j] = 0;
            if (sc[j] < 0.5f) continue;
            int p = id[j];
            int r = p / WI, cc = p - r * WI;
            int pr = r + 4, pc = cc + 4;
            if ((mask[pr * 25 + (pc >> 5)] >> (pc & 31)) & 1u) {
                for (int dr = -4; dr <= 4; ++dr) {
                    int rr = pr + dr;
                    for (int dc = -4; dc <= 4; ++dc) {
                        int c2 = pc + dc;
                        mask[rr * 25 + (c2 >> 5)] &= ~(1u << (c2 & 31));
                    }
                }
                int bord = 8;
                bool inb = (pr > bord) && (pr <= WI - bord) &&
                           (pc > bord) && (pc <= HI - bord);
                keep[j] = inb ? 1 : 0;
            }
        }
        float* pts   = out;
        float* confs = out + (size_t)BATCH * KTOP * 2;
        float* valid = confs + (size_t)BATCH * KTOP;
        int s = 0;
        for (int pass = 0; pass < 2; ++pass) {
            for (int j = 0; j < KTOP; ++j) {
                if ((keep[j] != 0) == (pass == 0)) {
                    int p = id[j];
                    pts[((size_t)img * KTOP + s) * 2 + 0] = (float)(p / WI);
                    pts[((size_t)img * KTOP + s) * 2 + 1] = (float)(p % WI);
                    confs[(size_t)img * KTOP + s] = sc[j];
                    valid[(size_t)img * KTOP + s] = keep[j] ? 1.f : 0.f;
                    ++s;
                }
            }
        }
    }
}

// ---------------------------------------------------------------------------
extern "C" void kernel_launch(void* const* d_in, const int* in_sizes, int n_in,
                              void* d_out, int out_size, void* d_ws, size_t ws_size,
                              hipStream_t stream)
{
    const float* x  = (const float*)d_in[0];
    const float* w1 = (const float*)d_in[1];
    const float* b1 = (const float*)d_in[2];
    const float* w2 = (const float*)d_in[3];
    const float* b2 = (const float*)d_in[4];
    float* out = (float*)d_out;

    char* ws = (char*)d_ws;
    const size_t w1hB  = (size_t)CMID * K1 * sizeof(_Float16);            //  0.6 MB
    const size_t w2hB  = (size_t)MPAD * CMID * sizeof(_Float16);          //  40 KB
    const size_t xhB   = (size_t)BATCH * CIN * PADP * sizeof(_Float16);   // 39.3 MB
    const size_t hB    = (size_t)BATCH * NPIX * CMID * sizeof(_Float16);  // 75.5 MB
    const size_t heatB = (size_t)BATCH * HI * WI * sizeof(float);         // 37.7 MB

    _Float16* w1h  = (_Float16*)(ws);
    _Float16* w2h  = (_Float16*)(ws + w1hB);
    _Float16* xh   = (_Float16*)(ws + w1hB + w2hB);
    _Float16* h    = (_Float16*)(ws + w1hB + w2hB + xhB);
    float*    heat = (float*)   (ws + w1hB + w2hB + xhB + hB);
    int*      cnt  = (int*)     (ws + w1hB + w2hB + xhB + hB + heatB);
    float2*   cand = (float2*)  (ws + w1hB + w2hB + xhB + hB + heatB + 256);

    // One-time f16 conversions (w1 keeps its k = ci*9 + tap contiguity).
    sp_cvt_f32_to_f16<<<(CMID * K1 / 4 + 255) / 256, 256, 0, stream>>>(w1, w1h, CMID * K1 / 4);
    sp_cvt_w2<<<(MPAD * CMID / 4 + 255) / 256, 256, 0, stream>>>(w2, w2h);
    {
        unsigned total = (unsigned)BATCH * CIN * PADP;
        sp_cvt_x_pad<<<(total + 255) / 256, 256, 0, stream>>>(x, xh);
    }
    sp_zero_cnt<<<1, 32, 0, stream>>>(cnt);

    // conv1: grid = (9216/32 pixel tiles, 256/128 chan tiles, 16 images)
    sp_conv1_wmma<<<dim3(NPIX / 32, CMID / 128, BATCH), 256, 0, stream>>>(xh, w1h, b1, h);

    // conv2 + softmax: grid = (9216/16 pixel tiles, 16 images)
    sp_conv2_softmax<<<dim3(NPIX / 16, BATCH), 160, 0, stream>>>(h, w2h, b2, heat);

    sp_collect<<<dim3((HI * WI + 255) / 256, BATCH), 256, 0, stream>>>(heat, cnt, cand);

    // dynamic LDS: 16384 + 16384 + 77600 + 4096 = 114464 B (CDNA5: 320KB/WG)
    sp_nms<<<BATCH, 1024, 114464, stream>>>(heat, cnt, cand, out);
}
